// STAN_86079734546499
// MI455X (gfx1250) — compile-verified, matrix operand order checked by reading, not run on hardware
//
#include <hip/hip_runtime.h>
#include <hip/hip_bf16.h>
#include <math.h>

typedef __bf16 bf16_t;
typedef __attribute__((ext_vector_type(16))) __bf16 v16bf;
typedef __attribute__((ext_vector_type(8)))  __bf16 v8bf;
typedef __attribute__((ext_vector_type(8)))  float  v8f;

// ---------------------------------------------------------------------------
// WMMA fragment layout (CDNA5 ISA 7.12.2, 16-bit A 16x32 / B 32x16):
//  lane L<16  : row/col = L,    kb = 0 ; lane L>=16 : row/col = L-16, kb = 8
//  fragment element i: K = kb + i        (i = 0..7)
//                      K = kb + 8 + i    (i = 8..15)  i.e. kb+16 .. kb+23
// C/D f32 16x16: lane -> col (lane&15), VGPR j -> row j (+8 if lane>=16).
// ---------------------------------------------------------------------------

// ============================================================
// Layer-1 GEMM: z[N,32] = bf16(Xf) @ bf16(W1^T) + b1
// Xf[n, t*4+f] = X[t*N*4 + n*4 + f]   (X is [1,16,N,4])
// One wave per 16-row tile; 2 col tiles x 2 K-steps of v_wmma_f32_16x16x32_bf16.
// Row index is CLAMPED (not predicated) so all fragment loads are
// unconditional 128-bit loads — no exec-mask churn around the WMMAs.
// ============================================================
__global__ void stan_gat1_gemm(const float* __restrict__ X,
                               const float* __restrict__ W1,
                               const float* __restrict__ b1,
                               float* __restrict__ z, int N) {
    int lane = threadIdx.x & 31;
    int tile = blockIdx.x * (blockDim.x >> 5) + (threadIdx.x >> 5);
    if (tile * 16 >= N) return;                 // wave-uniform exit: EXEC stays full
    const int row0 = tile * 16;
    const int rowA = row0 + (lane & 15);
    const int rowC = (rowA < N) ? rowA : (N - 1);   // clamp: rows >= N never stored
    const int kb   = (lane & 16) ? 8 : 0;
    const int col  = lane & 15;

    const float4* X4 = (const float4*)X;        // X4[t*N + n] = X[:, t, n, 0:4]
    const float4* W4 = (const float4*)W1;       // W4[c*16 + k/4] = W1[c, 4k:4k+4]

    v8f acc0 = {}; v8f acc1 = {};
#pragma unroll
    for (int kk = 0; kk < 64; kk += 32) {
        // ---- A fragment: elements 0..7 -> K = kk+kb+{0..7} (t = tb, tb+1)
        //                  elements 8..15 -> K = kk+kb+16+{0..7} (t = tb+4, tb+5)
        const int tb = (kk + kb) >> 2;
        float4 p0 = X4[(size_t)(tb + 0) * N + rowC];
        float4 p1 = X4[(size_t)(tb + 1) * N + rowC];
        float4 p2 = X4[(size_t)(tb + 4) * N + rowC];
        float4 p3 = X4[(size_t)(tb + 5) * N + rowC];
        v16bf a;
        a[0]  = (bf16_t)p0.x; a[1]  = (bf16_t)p0.y; a[2]  = (bf16_t)p0.z; a[3]  = (bf16_t)p0.w;
        a[4]  = (bf16_t)p1.x; a[5]  = (bf16_t)p1.y; a[6]  = (bf16_t)p1.z; a[7]  = (bf16_t)p1.w;
        a[8]  = (bf16_t)p2.x; a[9]  = (bf16_t)p2.y; a[10] = (bf16_t)p2.z; a[11] = (bf16_t)p2.w;
        a[12] = (bf16_t)p3.x; a[13] = (bf16_t)p3.y; a[14] = (bf16_t)p3.z; a[15] = (bf16_t)p3.w;

        // ---- B fragments for col tiles 0 and 1 (weight rows, contiguous K)
        const int k4 = (kk + kb) >> 2;
#pragma unroll
        for (int ct = 0; ct < 2; ++ct) {
            const int c = ct * 16 + col;
            float4 q0 = W4[(size_t)c * 16 + k4];
            float4 q1 = W4[(size_t)c * 16 + k4 + 1];
            float4 q2 = W4[(size_t)c * 16 + k4 + 4];
            float4 q3 = W4[(size_t)c * 16 + k4 + 5];
            v16bf b;
            b[0]  = (bf16_t)q0.x; b[1]  = (bf16_t)q0.y; b[2]  = (bf16_t)q0.z; b[3]  = (bf16_t)q0.w;
            b[4]  = (bf16_t)q1.x; b[5]  = (bf16_t)q1.y; b[6]  = (bf16_t)q1.z; b[7]  = (bf16_t)q1.w;
            b[8]  = (bf16_t)q2.x; b[9]  = (bf16_t)q2.y; b[10] = (bf16_t)q2.z; b[11] = (bf16_t)q2.w;
            b[12] = (bf16_t)q3.x; b[13] = (bf16_t)q3.y; b[14] = (bf16_t)q3.z; b[15] = (bf16_t)q3.w;
            if (ct == 0)
                acc0 = __builtin_amdgcn_wmma_f32_16x16x32_bf16(false, a, false, b, (short)0, acc0, false, false);
            else
                acc1 = __builtin_amdgcn_wmma_f32_16x16x32_bf16(false, a, false, b, (short)0, acc1, false, false);
        }
    }
    const int rbase = row0 + ((lane & 16) ? 8 : 0);
    const float bv0 = b1[col], bv1 = b1[col + 16];
#pragma unroll
    for (int j = 0; j < 8; ++j) {
        int r = rbase + j;
        if (r < N) {
            z[(size_t)r * 32 + col]      = acc0[j] + bv0;
            z[(size_t)r * 32 + 16 + col] = acc1[j] + bv1;
        }
    }
}

// ============================================================
// Generic node GEMM: out[N,OUTD] = A(bf16,[N,32]) @ W^T + bias, K=32.
// A rows are 64B: the two 8-element K-runs per lane are 16B-aligned v8bf loads.
// Used for GAT layer 2 (OUTD=32) and GRU input gates (OUTD=96).
// ============================================================
template <int OUTD>
__global__ void stan_node_gemm(const bf16_t* __restrict__ A,
                               const float* __restrict__ W,
                               const float* __restrict__ bias,
                               float* __restrict__ out, int N) {
    int lane = threadIdx.x & 31;
    int tile = blockIdx.x * (blockDim.x >> 5) + (threadIdx.x >> 5);
    if (tile * 16 >= N) return;
    const int row0 = tile * 16;
    const int rowA = row0 + (lane & 15);
    const int rowC = (rowA < N) ? rowA : (N - 1);
    const int kb   = (lane & 16) ? 8 : 0;
    const int col0 = lane & 15;

    v8bf lo = *(const v8bf*)(A + (size_t)rowC * 32 + kb);        // K = kb..kb+7
    v8bf hi = *(const v8bf*)(A + (size_t)rowC * 32 + kb + 16);   // K = kb+16..kb+23
    v16bf a;
#pragma unroll
    for (int i = 0; i < 8; ++i) { a[i] = lo[i]; a[8 + i] = hi[i]; }

    const float4* W4 = (const float4*)W;        // W4[c*8 + k/4] = W[c, 4k:4k+4]
    const int rbase = row0 + ((lane & 16) ? 8 : 0);
#pragma unroll
    for (int ct = 0; ct < OUTD / 16; ++ct) {
        const int c = ct * 16 + col0;
        float4 q0 = W4[(size_t)c * 8 + (kb >> 2)];
        float4 q1 = W4[(size_t)c * 8 + (kb >> 2) + 1];
        float4 q2 = W4[(size_t)c * 8 + (kb >> 2) + 4];
        float4 q3 = W4[(size_t)c * 8 + (kb >> 2) + 5];
        v16bf b;
        b[0]  = (bf16_t)q0.x; b[1]  = (bf16_t)q0.y; b[2]  = (bf16_t)q0.z; b[3]  = (bf16_t)q0.w;
        b[4]  = (bf16_t)q1.x; b[5]  = (bf16_t)q1.y; b[6]  = (bf16_t)q1.z; b[7]  = (bf16_t)q1.w;
        b[8]  = (bf16_t)q2.x; b[9]  = (bf16_t)q2.y; b[10] = (bf16_t)q2.z; b[11] = (bf16_t)q2.w;
        b[12] = (bf16_t)q3.x; b[13] = (bf16_t)q3.y; b[14] = (bf16_t)q3.z; b[15] = (bf16_t)q3.w;
        v8f cacc = {};
        cacc = __builtin_amdgcn_wmma_f32_16x16x32_bf16(false, a, false, b, (short)0, cacc, false, false);
        const float bv = bias[c];
#pragma unroll
        for (int j = 0; j < 8; ++j) {
            int r = rbase + j;
            if (r < N) out[(size_t)r * OUTD + c] = cacc[j] + bv;
        }
    }
}

// ============================================================
// Per-node attention scores: sS[n] = aW[0:32]·z[n], sD[n] = aW[32:64]·z[n]
// ============================================================
__global__ void stan_node_scores(const float* __restrict__ z,
                                 const float* __restrict__ aW,
                                 float* __restrict__ sS, float* __restrict__ sD, int N) {
    int n = blockIdx.x * blockDim.x + threadIdx.x;
    if (n >= N) return;
    float a = 0.f, b = 0.f;
    const float* zr = z + (size_t)n * 32;
#pragma unroll
    for (int k = 0; k < 32; ++k) { a += aW[k] * zr[k]; b += aW[32 + k] * zr[k]; }
    sS[n] = a; sD[n] = b;
}

// ============================================================
// Edge kernel: one edge per wave32; lane k scatters att*z[src][k] -> acc[dst][k]
// att = sigmoid(leaky_relu(sS[src] + sD[dst] + ab, 0.01))
// z/acc (6.4 MB each) are L2-resident on MI455X (192 MB L2): atomics stay on-chip.
// ============================================================
__global__ void stan_edge_scatter(const int* __restrict__ adj, int E,
                                  const float* __restrict__ z,
                                  const float* __restrict__ sS,
                                  const float* __restrict__ sD,
                                  const float* __restrict__ abp,
                                  float* __restrict__ acc) {
    int e    = blockIdx.x * (blockDim.x >> 5) + (threadIdx.x >> 5);
    int lane = threadIdx.x & 31;
    if (e >= E) return;
    int src = adj[e];
    int dst = adj[E + e];
    float ee = sS[src] + sD[dst] + abp[0];
    float lr = (ee > 0.f) ? ee : 0.01f * ee;
    float att = 1.f / (1.f + __expf(-lr));
    float v = att * z[(size_t)src * 32 + lane];
    atomicAdd(&acc[(size_t)dst * 32 + lane], v);
}

// ============================================================
// ELU then cast to bf16 (feeds next WMMA GEMM)
// ============================================================
__global__ void stan_elu_bf16(const float* __restrict__ acc,
                              bf16_t* __restrict__ hbf, int total) {
    int i = blockIdx.x * blockDim.x + threadIdx.x;
    if (i >= total) return;
    float x = acc[i];
    float y = (x > 0.f) ? x : (__expf(x) - 1.f);
    hbf[i] = (bf16_t)y;
}

// ============================================================
// GRU gates (h0 = 0), prediction heads, SIR physics scan. One thread per node.
// ============================================================
__global__ void stan_gru_heads_sir(const float* __restrict__ gi,
                                   const float* __restrict__ b_hh,
                                   const float* __restrict__ X,
                                   const float* __restrict__ states,
                                   const float* __restrict__ Wi, const float* __restrict__ bi,
                                   const float* __restrict__ Wr, const float* __restrict__ br,
                                   const float* __restrict__ Ws, const float* __restrict__ bs,
                                   float* __restrict__ outPred, float* __restrict__ outPhy,
                                   int N) {
    int n = blockIdx.x * blockDim.x + threadIdx.x;
    if (n >= N) return;
    float hc[34];
    const float* g = gi + (size_t)n * 96;
#pragma unroll
    for (int k = 0; k < 32; ++k) {
        float r  = 1.f / (1.f + __expf(-(g[k]      + b_hh[k])));
        float zg = 1.f / (1.f + __expf(-(g[32 + k] + b_hh[32 + k])));
        float nn = tanhf(g[64 + k] + r * b_hh[64 + k]);
        hc[k] = (1.f - zg) * nn;                 // + zg*h0, h0 = 0
    }
    hc[32] = X[((size_t)15 * N + n) * 4 + 1];    // last_diff_I = X[:, -1, n, 1]
    hc[33] = X[((size_t)15 * N + n) * 4 + 2];    // last_diff_R = X[:, -1, n, 2]

    float* po = outPred + (size_t)n * 16;
#pragma unroll
    for (int t = 0; t < 8; ++t) {
        float pi = bi[t], pr = br[t];
        for (int k = 0; k < 34; ++k) { pi += Wi[t * 34 + k] * hc[k]; pr += Wr[t * 34 + k] * hc[k]; }
        po[t * 2]     = pi;
        po[t * 2 + 1] = pr;
    }
    float a0 = bs[0], b0 = bs[1];
    for (int k = 0; k < 34; ++k) { a0 += Ws[k] * hc[k]; b0 += Ws[34 + k] * hc[k]; }
    float alpha = 1.f / (1.f + __expf(-a0));
    float beta  = 1.f / (1.f + __expf(-b0));

    float I = states[(size_t)n * 2], R = states[(size_t)n * 2 + 1];
    float* ph = outPhy + (size_t)n * 16;
#pragma unroll
    for (int t = 0; t < 8; ++t) {
        float S  = 1e10f - I - R;
        float dI = alpha * I * (S * 1e-10f) - beta * I;
        float dR = beta * I;
        ph[t * 2]     = dI;
        ph[t * 2 + 1] = dR;
        I += dI; R += dR;                        // stop_gradient = identity in fwd
    }
}

// ============================================================
// Host launcher
// ============================================================
extern "C" void kernel_launch(void* const* d_in, const int* in_sizes, int n_in,
                              void* d_out, int out_size, void* d_ws, size_t ws_size,
                              hipStream_t stream) {
    const float* X      = (const float*)d_in[0];
    const int*   adj    = (const int*)  d_in[1];
    const float* states = (const float*)d_in[2];
    const float* W1     = (const float*)d_in[3];
    const float* b1     = (const float*)d_in[4];
    const float* aW1    = (const float*)d_in[5];
    const float* ab1    = (const float*)d_in[6];
    const float* W2     = (const float*)d_in[7];
    const float* b2     = (const float*)d_in[8];
    const float* aW2    = (const float*)d_in[9];
    const float* ab2    = (const float*)d_in[10];
    const float* W_ih   = (const float*)d_in[11];
    /* W_hh = d_in[12] unused: h0 == 0 */
    const float* b_ih   = (const float*)d_in[13];
    const float* b_hh   = (const float*)d_in[14];
    const float* Wi     = (const float*)d_in[15];
    const float* bi     = (const float*)d_in[16];
    const float* Wr     = (const float*)d_in[17];
    const float* br     = (const float*)d_in[18];
    const float* Ws     = (const float*)d_in[19];
    const float* bs     = (const float*)d_in[20];

    const int N = in_sizes[2] / 2;   // states [N,2]
    const int E = in_sizes[1] / 2;   // adj [2,E]

    // Workspace carve-up (~36 MB for N=50k)
    float*  z   = (float*)d_ws;                  // N*32
    float*  acc = z   + (size_t)N * 32;          // N*32 (atomic target)
    float*  sS  = acc + (size_t)N * 32;          // N
    float*  sD  = sS  + N;                       // N
    float*  gi  = sD  + N;                       // N*96
    bf16_t* hbf = (bf16_t*)(gi + (size_t)N * 96);// N*32 bf16

    const int BLK = 256;
    const int wavesPerBlk = BLK / 32;
    const int tiles     = (N + 15) / 16;
    const int gemmGrid  = (tiles + wavesPerBlk - 1) / wavesPerBlk;
    const int nodeGrid  = (N + BLK - 1) / BLK;
    const int elemGrid  = (N * 32 + BLK - 1) / BLK;
    const int edgeGrid  = (E + wavesPerBlk - 1) / wavesPerBlk;

    // ---- GAT layer 1 ----
    hipMemsetAsync(acc, 0, (size_t)N * 32 * sizeof(float), stream);
    stan_gat1_gemm<<<gemmGrid, BLK, 0, stream>>>(X, W1, b1, z, N);
    stan_node_scores<<<nodeGrid, BLK, 0, stream>>>(z, aW1, sS, sD, N);
    stan_edge_scatter<<<edgeGrid, BLK, 0, stream>>>(adj, E, z, sS, sD, ab1, acc);
    stan_elu_bf16<<<elemGrid, BLK, 0, stream>>>(acc, hbf, N * 32);

    // ---- GAT layer 2 ----
    hipMemsetAsync(acc, 0, (size_t)N * 32 * sizeof(float), stream);
    stan_node_gemm<32><<<gemmGrid, BLK, 0, stream>>>(hbf, W2, b2, z, N);
    stan_node_scores<<<nodeGrid, BLK, 0, stream>>>(z, aW2, sS, sD, N);
    stan_edge_scatter<<<edgeGrid, BLK, 0, stream>>>(adj, E, z, sS, sD, ab2, acc);
    stan_elu_bf16<<<elemGrid, BLK, 0, stream>>>(acc, hbf, N * 32);

    // ---- GRU input gates (WMMA) + fused gates/heads/SIR ----
    stan_node_gemm<96><<<gemmGrid, BLK, 0, stream>>>(hbf, W_ih, b_ih, gi, N);
    stan_gru_heads_sir<<<nodeGrid, BLK, 0, stream>>>(gi, b_hh, X, states,
                                                     Wi, bi, Wr, br, Ws, bs,
                                                     (float*)d_out,
                                                     (float*)d_out + (size_t)N * 16, N);
}
